// EGA_54803782696976
// MI455X (gfx1250) — compile-verified, matrix operand order checked by reading, not run on hardware
//
#include <hip/hip_runtime.h>
#include <math.h>

// ---------------------------------------------------------------------------
// Types for gfx1250 WMMA (wave32): v_wmma_f32_16x16x32_bf16
// ---------------------------------------------------------------------------
typedef __attribute__((ext_vector_type(8)))  float          v8f;
typedef __attribute__((ext_vector_type(16))) __bf16         v16bf;
typedef __attribute__((ext_vector_type(8)))  __bf16         v8bf;
typedef __attribute__((ext_vector_type(4)))  unsigned int   u32x4;
typedef __attribute__((ext_vector_type(4)))  float          f32x4;
typedef __attribute__((ext_vector_type(4)))  unsigned short u16x4;
typedef __attribute__((address_space(3)))    unsigned short lds_u16;

// Problem dims (fixed by the reference)
static constexpr int   Bn = 4, Tn = 4096, Fn = 1024, Hn = 16, Dn = 64, TDn = 1024;
static constexpr float EPSv  = 1e-5f;
static constexpr float SCALE = 0.125f;   // 1/sqrt(D)

static constexpr int EPI_SCORES = 0;  // Cf = alpha*acc
static constexpr int EPI_ADD    = 1;  // Cf += alpha*acc
static constexpr int EPI_QK     = 2;  // bf16 scatter to [b,h,t,d], +bias
static constexpr int EPI_VT     = 3;  // bf16 scatter to [b,h,d,t], +bias
static constexpr int EPI_CTX    = 4;  // bf16 scatter to [b,t,h*64+d]
static constexpr int EPI_OUT    = 5;  // f32 (acc+bias)*ls_gamma
static constexpr int EPI_GATE   = 6;  // f32 sigmoid(acc+bias)*attn_up

__device__ __forceinline__ unsigned short f2bf(float f) {
  unsigned int u = __float_as_uint(f);
  u += 0x7FFFu + ((u >> 16) & 1u);           // round-to-nearest-even
  return (unsigned short)(u >> 16);
}

__device__ __forceinline__ v16bf cat8(v8bf lo, v8bf hi) {
  return __builtin_shufflevector(lo, hi, 0,1,2,3,4,5,6,7,8,9,10,11,12,13,14,15);
}

__device__ __forceinline__ float waveSum(float v) {
#pragma unroll
  for (int m = 16; m; m >>= 1) v += __shfl_xor(v, m, 32);
  return v;
}
__device__ __forceinline__ float waveMax(float v) {
#pragma unroll
  for (int m = 16; m; m >>= 1) v = fmaxf(v, __shfl_xor(v, m, 32));
  return v;
}

// ---------------------------------------------------------------------------
// Canonical batched GEMM: C[m][n] = sum_k A[m][k] * Bt[n][k]  (both bf16/u16)
// Block = 128 threads (4 waves), tile 64x64, K-step 32, double-buffered LDS
// filled with GLOBAL_LOAD_ASYNC_TO_LDS_B128 (ASYNCcnt-tracked, no VGPR bounce).
// Requires K % 64 == 0 (true for all uses: K = 64 or 1024).
// Each wave: 32x32 via 2x2 v_wmma_f32_16x16x32_bf16.
// LDS rows padded to 40 u16 (80B = 20 banks) -> conflict-free b128 frag reads.
// ---------------------------------------------------------------------------
struct GemmParams {
  const unsigned short* A;  long lda, bsA;
  const unsigned short* Bt; long ldb, bsB;
  int M, N, K;
  float alpha;
  const float* bias;     // len N (or unused)
  float* Cf; long ldc, bsC;
  unsigned short* Cb;    // bf16 output target
  const float* lsg;      // layer-scale gamma
  const float* gateSrc;  // attn output (downsampled) for gate epilogue
  float* outF;           // final output
};

template <int EPI>
__global__ __launch_bounds__(128)
void gemm_bf16(GemmParams p)
{
  __shared__ __align__(16) unsigned short As0[64 * 40], Bs0[64 * 40];
  __shared__ __align__(16) unsigned short As1[64 * 40], Bs1[64 * 40];

  const int z    = blockIdx.z;
  const int m0   = blockIdx.y * 64;
  const int n0   = blockIdx.x * 64;
  const int tid  = threadIdx.x;
  const int lane = tid & 31;
  const int wave = tid >> 5;
  const int wm   = (wave >> 1) * 32;   // wave origin in block tile (rows)
  const int wn   = (wave & 1) * 32;    // wave origin in block tile (cols)
  const int half = lane >> 4;          // K-half selector per WMMA layout
  const int lr   = lane & 15;          // row/col within 16-tile
  const int cc   = half * 8;           // K chunk offset (0 or 8)

  // Staging map: thread stages rows sr and sr+32, 16B chunk scol.
  const int sr   = tid >> 2;
  const int scol = (tid & 3) * 8;
  const int st0  = sr * 40 + scol;
  const int st1  = (sr + 32) * 40 + scol;

  const unsigned short* aP0 = p.A  + (long)z * p.bsA + (long)(m0 + sr) * p.lda + scol;
  const unsigned short* aP1 = aP0 + 32 * p.lda;
  const unsigned short* bP0 = p.Bt + (long)z * p.bsB + (long)(n0 + sr) * p.ldb + scol;
  const unsigned short* bP1 = bP0 + 32 * p.ldb;

  // LDS destinations (addrspace(3) -> 32-bit VGPR operand for async-to-LDS)
  lds_u16* dA0_0 = (lds_u16*)&As0[st0];
  lds_u16* dA1_0 = (lds_u16*)&As0[st1];
  lds_u16* dB0_0 = (lds_u16*)&Bs0[st0];
  lds_u16* dB1_0 = (lds_u16*)&Bs0[st1];
  lds_u16* dA0_1 = (lds_u16*)&As1[st0];
  lds_u16* dA1_1 = (lds_u16*)&As1[st1];
  lds_u16* dB0_1 = (lds_u16*)&Bs1[st0];
  lds_u16* dB1_1 = (lds_u16*)&Bs1[st1];

  // Fragment LDS offsets (constant per thread; buffer base alternates statically)
  const int fA0 = (wm + lr) * 40 + cc;
  const int fA1 = (wm + 16 + lr) * 40 + cc;
  const int fB0 = (wn + lr) * 40 + cc;
  const int fB1 = (wn + 16 + lr) * 40 + cc;

  const v8f vzero = {0.f, 0.f, 0.f, 0.f, 0.f, 0.f, 0.f, 0.f};
  v8f acc[2][2] = {{vzero, vzero}, {vzero, vzero}};

  // Async DMA of one 64x32 A tile + 64x32 B tile into the given buffer's slots.
  auto asyncFetch = [&](lds_u16* a0d, lds_u16* a1d, lds_u16* b0d, lds_u16* b1d) {
    asm volatile(
        "global_load_async_to_lds_b128 %0, %4, off\n\t"
        "global_load_async_to_lds_b128 %1, %5, off\n\t"
        "global_load_async_to_lds_b128 %2, %6, off\n\t"
        "global_load_async_to_lds_b128 %3, %7, off"
        :
        : "v"(a0d), "v"(a1d), "v"(b0d), "v"(b1d),
          "v"(aP0), "v"(aP1), "v"(bP0), "v"(bP1)
        : "memory");
    aP0 += 32; aP1 += 32; bP0 += 32; bP1 += 32;
  };
  auto waitAsync = [] {
    asm volatile("s_wait_asynccnt 0x0" ::: "memory");
  };

  auto compute = [&](const unsigned short* Asrc, const unsigned short* Bsrc) {
    const v16bf a0 = cat8(*(const v8bf*)&Asrc[fA0], *(const v8bf*)&Asrc[fA0 + 16]);
    const v16bf a1 = cat8(*(const v8bf*)&Asrc[fA1], *(const v8bf*)&Asrc[fA1 + 16]);
    const v16bf b0 = cat8(*(const v8bf*)&Bsrc[fB0], *(const v8bf*)&Bsrc[fB0 + 16]);
    const v16bf b1 = cat8(*(const v8bf*)&Bsrc[fB1], *(const v8bf*)&Bsrc[fB1 + 16]);
    acc[0][0] = __builtin_amdgcn_wmma_f32_16x16x32_bf16(false, a0, false, b0, (short)0, acc[0][0], false, false);
    acc[0][1] = __builtin_amdgcn_wmma_f32_16x16x32_bf16(false, a0, false, b1, (short)0, acc[0][1], false, false);
    acc[1][0] = __builtin_amdgcn_wmma_f32_16x16x32_bf16(false, a1, false, b0, (short)0, acc[1][0], false, false);
    acc[1][1] = __builtin_amdgcn_wmma_f32_16x16x32_bf16(false, a1, false, b1, (short)0, acc[1][1], false, false);
  };

  // Prologue: tile 0 -> buf0
  asyncFetch(dA0_0, dA1_0, dB0_0, dB1_0);
  waitAsync();
  __syncthreads();

  // Main loop: K % 64 == 0, two 32-wide phases with static double buffers.
  // Async loads for buf^1 are issued before computing buf (prev barrier ensures
  // all waves finished reading buf^1); each wave waits for its own ASYNCcnt,
  // then the barrier publishes the buffer to the workgroup.
  for (int k0 = 0;;) {
    // phase 0: buf0 holds tile k0; DMA tile k0+32 into buf1 (always exists)
    asyncFetch(dA0_1, dA1_1, dB0_1, dB1_1);
    compute(As0, Bs0);
    waitAsync();
    __syncthreads();
    k0 += 32;

    // phase 1: buf1 holds tile k0; DMA tile k0+32 into buf0 unless done
    const bool last = (k0 + 32 >= p.K);
    if (!last) asyncFetch(dA0_0, dA1_0, dB0_0, dB1_0);
    compute(As1, Bs1);
    if (last) break;
    waitAsync();
    __syncthreads();
    k0 += 32;
  }

  // Epilogue: C 16x16 f32 layout -> n = lr, m = half*8 + v
#pragma unroll
  for (int i = 0; i < 2; ++i) {
#pragma unroll
    for (int j = 0; j < 2; ++j) {
      const int n  = n0 + wn + j * 16 + lr;
      const int mB = m0 + wm + i * 16 + half * 8;
#pragma unroll
      for (int v = 0; v < 8; ++v) {
        const int m = mB + v;
        const float val = p.alpha * acc[i][j][v];
        if constexpr (EPI == EPI_SCORES) {
          p.Cf[(long)z * p.bsC + (long)m * p.ldc + n] = val;
        } else if constexpr (EPI == EPI_ADD) {
          p.Cf[(long)z * p.bsC + (long)m * p.ldc + n] += val;
        } else if constexpr (EPI == EPI_QK) {
          const int b = m >> 10, t = m & 1023, h = n >> 6, d = n & 63;
          p.Cb[(((long)(b * Hn + h) * TDn + t) << 6) + d] = f2bf(val + p.bias[n]);
        } else if constexpr (EPI == EPI_VT) {
          const int b = m >> 10, t = m & 1023, h = n >> 6, d = n & 63;
          p.Cb[(((long)(b * Hn + h) * Dn + d) << 10) + t] = f2bf(val + p.bias[n]);
        } else if constexpr (EPI == EPI_CTX) {
          const int b = z >> 4, h = z & 15;
          p.Cb[(((long)b * TDn + m) << 10) + h * Dn + n] = f2bf(val);
        } else if constexpr (EPI == EPI_OUT) {
          p.Cf[((long)m << 10) + n] = (val + p.bias[n]) * p.lsg[n];
        } else {  // EPI_GATE: m = b*4096 + t ; attn row = m>>2 ; nearest upsample
          const float gv = 1.f / (1.f + __expf(-(val + p.bias[n])));
          p.outF[((long)m << 10) + n] = gv * p.gateSrc[((long)(m >> 2) << 10) + n];
        }
      }
    }
  }
}

// ---------------------------------------------------------------------------
// Prep kernels
// ---------------------------------------------------------------------------
__global__ __launch_bounds__(256)
void k_conv_w(const float* s0, const float* s1, const float* s2, const float* s3,
              const float* s4, unsigned short* d0, unsigned short* d1,
              unsigned short* d2, unsigned short* d3, unsigned short* d4)
{
  const float* S[5] = {s0, s1, s2, s3, s4};
  unsigned short* D[5] = {d0, d1, d2, d3, d4};
  const long stride = (long)gridDim.x * blockDim.x;
  for (long i = (long)blockIdx.x * blockDim.x + threadIdx.x; i < 5242880L; i += stride) {
    const int  w = (int)(i >> 20);
    const long e = i & 1048575L;           // e = k*1024 + n   (W is [k][n])
    const long k = e >> 10, n = e & 1023;
    D[w][n * 1024 + k] = f2bf(S[w][e]);    // store transposed Wt[n][k]
  }
}

__global__ __launch_bounds__(256)
void k_conv_pos(const float* __restrict__ s, unsigned short* __restrict__ d)
{
  const long stride = 4L * gridDim.x * blockDim.x;
  for (long i = 4L * ((long)blockIdx.x * blockDim.x + threadIdx.x); i < 67108864L; i += stride) {
    const f32x4 v = *(const f32x4*)&s[i];
    u16x4 o;
    o.x = f2bf(v.x); o.y = f2bf(v.y); o.z = f2bf(v.z); o.w = f2bf(v.w);
    *(u16x4*)&d[i] = o;
  }
}

__global__ __launch_bounds__(256)
void k_down(const float* __restrict__ x, unsigned short* __restrict__ xd)
{
  const long stride = (long)gridDim.x * blockDim.x;
  for (long i = (long)blockIdx.x * blockDim.x + threadIdx.x; i < 4194304L; i += stride) {
    const int f  = (int)(i & 1023);
    const int td = (int)((i >> 10) & 1023);
    const int b  = (int)(i >> 20);
    const float* base = x + ((long)b * Tn + td * 4) * 1024 + f;
    xd[i] = f2bf(0.25f * (base[0] + base[1024] + base[2048] + base[3072]));
  }
}

__global__ __launch_bounds__(256)
void k_ln(const float* __restrict__ x, const float* __restrict__ g,
          const float* __restrict__ b, unsigned short* __restrict__ y)
{
  const long row = blockIdx.x;
  const float* xr = x + row * 1024L;
  const int tid = threadIdx.x;
  float v[4]; float s = 0.f, s2 = 0.f;
#pragma unroll
  for (int i = 0; i < 4; ++i) { v[i] = xr[tid + i * 256]; s += v[i]; s2 += v[i] * v[i]; }
  s  = waveSum(s);
  s2 = waveSum(s2);
  __shared__ float rs[8], rq[8];
  if ((tid & 31) == 0) { rs[tid >> 5] = s; rq[tid >> 5] = s2; }
  __syncthreads();
  float ts = 0.f, tq = 0.f;
#pragma unroll
  for (int i = 0; i < 8; ++i) { ts += rs[i]; tq += rq[i]; }
  const float mean = ts * (1.f / 1024.f);
  const float var  = tq * (1.f / 1024.f) - mean * mean;
  const float rstd = rsqrtf(var + EPSv);
  unsigned short* yr = y + row * 1024L;
#pragma unroll
  for (int i = 0; i < 4; ++i) {
    const int c = tid + i * 256;
    yr[c] = f2bf((v[i] - mean) * rstd * g[c] + b[c]);
  }
}

// Softmax over 1024-wide rows; writes bf16 probs in-place into the f32 scores
// buffer (row-private region, all reads precede first barrier).
__global__ __launch_bounds__(256)
void k_softmax(float* sc)
{
  const long row = blockIdx.x;
  float* sr = sc + row * 1024L;
  unsigned short* dr = (unsigned short*)sc + row * 2048L;
  const int tid = threadIdx.x;
  float v[4];
#pragma unroll
  for (int i = 0; i < 4; ++i) v[i] = sr[tid + i * 256];
  float mx = fmaxf(fmaxf(v[0], v[1]), fmaxf(v[2], v[3]));
  mx = waveMax(mx);
  __shared__ float red[8];
  if ((tid & 31) == 0) red[tid >> 5] = mx;
  __syncthreads();
  float rowMax = red[0];
#pragma unroll
  for (int i = 1; i < 8; ++i) rowMax = fmaxf(rowMax, red[i]);
  __syncthreads();
  float e[4]; float s = 0.f;
#pragma unroll
  for (int i = 0; i < 4; ++i) { e[i] = __expf(v[i] - rowMax); s += e[i]; }
  s = waveSum(s);
  if ((tid & 31) == 0) red[tid >> 5] = s;
  __syncthreads();
  float tot = 0.f;
#pragma unroll
  for (int i = 0; i < 8; ++i) tot += red[i];
  const float inv = 1.f / tot;
#pragma unroll
  for (int i = 0; i < 4; ++i) dr[tid + i * 256] = f2bf(e[i] * inv);
}

// ---------------------------------------------------------------------------
// Launch
// ---------------------------------------------------------------------------
extern "C" void kernel_launch(void* const* d_in, const int* in_sizes, int n_in,
                              void* d_out, int out_size, void* d_ws, size_t ws_size,
                              hipStream_t stream)
{
  (void)in_sizes; (void)n_in; (void)out_size; (void)ws_size;
  const float* x   = (const float*)d_in[0];
  const float* pos = (const float*)d_in[1];
  const float* Wq  = (const float*)d_in[2];
  const float* bq  = (const float*)d_in[3];
  const float* Wk  = (const float*)d_in[4];
  const float* bk  = (const float*)d_in[5];
  const float* Wv  = (const float*)d_in[6];
  const float* bv  = (const float*)d_in[7];
  const float* Wo  = (const float*)d_in[8];
  const float* bo  = (const float*)d_in[9];
  const float* lsg = (const float*)d_in[10];
  const float* gng = (const float*)d_in[11];
  const float* gnb = (const float*)d_in[12];
  const float* gW  = (const float*)d_in[13];
  const float* gb  = (const float*)d_in[14];
  float* out = (float*)d_out;

  char* ws = (char*)d_ws;
  size_t off = 0;
  auto take = [&](size_t bytes) -> void* {
    void* p = ws + off;
    off = (off + bytes + 255) & ~(size_t)255;
    return p;
  };
  unsigned short* WTq = (unsigned short*)take(1048576ull * 2);
  unsigned short* WTk = (unsigned short*)take(1048576ull * 2);
  unsigned short* WTv = (unsigned short*)take(1048576ull * 2);
  unsigned short* WTo = (unsigned short*)take(1048576ull * 2);
  unsigned short* WTg = (unsigned short*)take(1048576ull * 2);
  unsigned short* xd  = (unsigned short*)take(4194304ull * 2);   // [b,td,f] bf16
  unsigned short* xln = (unsigned short*)take(16777216ull * 2);  // [b,t,f] bf16
  unsigned short* qb  = (unsigned short*)take(4194304ull * 2);   // [bh,t,d]
  unsigned short* kb  = (unsigned short*)take(4194304ull * 2);   // [bh,t,d]
  unsigned short* vT  = (unsigned short*)take(4194304ull * 2);   // [bh,d,t]
  unsigned short* Pb  = (unsigned short*)take(67108864ull * 2);  // pos bf16
  float*          sc  = (float*)take(67108864ull * 4);           // scores f32
  unsigned short* ctx = (unsigned short*)take(4194304ull * 2);   // [b,t,f]
  float*          ao  = (float*)take(4194304ull * 4);            // attn out f32

  k_conv_w  <<<2048,  256, 0, stream>>>(Wq, Wk, Wv, Wo, gW, WTq, WTk, WTv, WTo, WTg);
  k_conv_pos<<<8192,  256, 0, stream>>>(pos, Pb);
  k_down    <<<4096,  256, 0, stream>>>(x, xd);
  k_ln      <<<16384, 256, 0, stream>>>(x, gng, gnb, xln);

  {  // QKV projections: M=4096 N=1024 K=1024
    GemmParams p{};
    p.A = xd; p.lda = 1024; p.bsA = 0;
    p.ldb = 1024; p.bsB = 0;
    p.M = 4096; p.N = 1024; p.K = 1024; p.alpha = 1.f;
    dim3 g(16, 64, 1);
    p.Bt = WTq; p.bias = bq; p.Cb = qb; gemm_bf16<EPI_QK><<<g, 128, 0, stream>>>(p);
    p.Bt = WTk; p.bias = bk; p.Cb = kb; gemm_bf16<EPI_QK><<<g, 128, 0, stream>>>(p);
    p.Bt = WTv; p.bias = bv; p.Cb = vT; gemm_bf16<EPI_VT><<<g, 128, 0, stream>>>(p);
  }
  {  // scores = scale * q k^T, batched over bh=64
    GemmParams p{};
    p.A = qb; p.lda = 64; p.bsA = 65536;
    p.Bt = kb; p.ldb = 64; p.bsB = 65536;
    p.M = 1024; p.N = 1024; p.K = 64; p.alpha = SCALE;
    p.Cf = sc; p.ldc = 1024; p.bsC = 1048576;
    gemm_bf16<EPI_SCORES><<<dim3(16, 16, 64), 128, 0, stream>>>(p);
  }
  {  // scores += scale * q[:, t, :] P[t]^T, batched over t=1024 (M=bh=64)
    GemmParams p{};
    p.A = qb; p.lda = 65536; p.bsA = 64;
    p.Bt = Pb; p.ldb = 64; p.bsB = 65536;
    p.M = 64; p.N = 1024; p.K = 64; p.alpha = SCALE;
    p.Cf = sc; p.ldc = 1048576; p.bsC = 1024;
    gemm_bf16<EPI_ADD><<<dim3(16, 1, 1024), 128, 0, stream>>>(p);
  }
  k_softmax<<<65536, 256, 0, stream>>>(sc);
  {  // ctx = attn_w @ v, batched over bh=64 (attn_w bf16 lives inside sc rows)
    GemmParams p{};
    p.A = (const unsigned short*)sc; p.lda = 2048; p.bsA = 2097152;
    p.Bt = vT; p.ldb = 1024; p.bsB = 65536;
    p.M = 1024; p.N = 64; p.K = 1024; p.alpha = 1.f;
    p.Cb = ctx;
    gemm_bf16<EPI_CTX><<<dim3(1, 16, 64), 128, 0, stream>>>(p);
  }
  {  // attn = (ctx @ Wo + bo) * ls_gamma
    GemmParams p{};
    p.A = ctx; p.lda = 1024; p.bsA = 0;
    p.Bt = WTo; p.ldb = 1024; p.bsB = 0;
    p.M = 4096; p.N = 1024; p.K = 1024; p.alpha = 1.f;
    p.bias = bo; p.lsg = lsg;
    p.Cf = ao; p.ldc = 1024; p.bsC = 0;
    gemm_bf16<EPI_OUT><<<dim3(16, 64, 1), 128, 0, stream>>>(p);
  }
  {  // out = sigmoid(xln @ gW + gb) * upsample(attn)
    GemmParams p{};
    p.A = xln; p.lda = 1024; p.bsA = 0;
    p.Bt = WTg; p.ldb = 1024; p.bsB = 0;
    p.M = 16384; p.N = 1024; p.K = 1024; p.alpha = 1.f;
    p.bias = gb; p.gateSrc = ao; p.outF = out;
    gemm_bf16<EPI_GATE><<<dim3(16, 256, 1), 128, 0, stream>>>(p);
  }
}
